// Self_Attention_87857851007669
// MI455X (gfx1250) — matrix operand order repaired and checked by reading, not compiled, not run on hardware
//
#include <hip/hip_runtime.h>

// ---------------------------------------------------------------------------
// Self-attention (no softmax, per reference) for MI455X / gfx1250.
//
//   Q = x Wq^T, K = x Wk^T, V = x Wv^T           (bf16 WMMA, f32 accumulate)
//   N = V^T K  (= (K^T V)^T, per batch, d x d)
//   out = (1/32) * Q @ (K^T V) = (1/32) * Q @ N^T (associativity rewrite; the
//   [b,s,s] score matrix is never materialized: 86 GF instead of 120 GF)
//
// Data path: bf16 operands staged global -> LDS with GLOBAL_LOAD_ASYNC_TO_LDS
// (ASYNCcnt) and double-buffered so the next K-panel overlaps the WMMAs.
// ---------------------------------------------------------------------------

typedef __bf16 bf16_t;
typedef __bf16 v16bf __attribute__((ext_vector_type(16)));
typedef __bf16 v8bf  __attribute__((ext_vector_type(8)));
typedef __bf16 v4bf  __attribute__((ext_vector_type(4)));
typedef float  v8f   __attribute__((ext_vector_type(8)));
typedef float  v4f   __attribute__((ext_vector_type(4)));
typedef int    v4i   __attribute__((vector_size(16)));  // matches builtin proto

typedef __attribute__((address_space(1))) v4i g_v4i;    // global int4
typedef __attribute__((address_space(3))) v4i l_v4i;    // LDS int4

#define M_BLK 128
#define N_BLK 128
#define K_BLK 64
#define LDT   72   // padded LDS row stride (bf16 elems): 144 B, keeps 16B align

#if __has_builtin(__builtin_amdgcn_global_load_async_to_lds_b128) && \
    __has_builtin(__builtin_amdgcn_s_wait_asynccnt)
#define HAVE_ASYNC_LDS 1
#else
#define HAVE_ASYNC_LDS 0
#endif

// A fragment, 16x32 bf16 (ISA 7.12.2): lane m<16 holds K = kk+hi*8 .. +7 in
// v0-3 and K = kk+16+hi*8 .. +7 in v4-7  -> two 16-byte LDS loads.
__device__ __forceinline__ v16bf frag_a(const bf16_t* s, int row, int kk, int hi) {
  const bf16_t* p = s + row * LDT + kk + hi * 8;
  v8bf lo = *(const v8bf*)p;
  v8bf hh = *(const v8bf*)(p + 16);
  return __builtin_shufflevector(lo, hh, 0,1,2,3,4,5,6,7,8,9,10,11,12,13,14,15);
}
// B fragment, 32x16 bf16: lane n holds column n of B; K halves split by lane group.
__device__ __forceinline__ v16bf frag_b(const bf16_t* s, int row, int kk, int hi) {
  const bf16_t* p = s + row * LDT + kk + hi * 8;
  v8bf lo = *(const v8bf*)p;
  v8bf hh = *(const v8bf*)(p + 16);
  return __builtin_shufflevector(lo, hh, 0,1,2,3,4,5,6,7,8,9,10,11,12,13,14,15);
}

// One K_BLK panel of WMMAs: wave computes a 32x64 sub-tile (2x4 16x16 tiles).
__device__ __forceinline__ void mma_block(const bf16_t* sA, const bf16_t* sB,
                                          v8f acc[2][4], int wm, int wn, int lane) {
  const int hi = lane >> 4;
  const int l  = lane & 15;
#pragma unroll
  for (int kk = 0; kk < K_BLK; kk += 32) {
    v16bf a[2], b[4];
#pragma unroll
    for (int mt = 0; mt < 2; ++mt) a[mt] = frag_a(sA, wm * 32 + mt * 16 + l, kk, hi);
#pragma unroll
    for (int nt = 0; nt < 4; ++nt) b[nt] = frag_b(sB, wn * 64 + nt * 16 + l, kk, hi);
#pragma unroll
    for (int mt = 0; mt < 2; ++mt)
#pragma unroll
      for (int nt = 0; nt < 4; ++nt)
        acc[mt][nt] = __builtin_amdgcn_wmma_f32_16x16x32_bf16(
            false, a[mt], false, b[nt], (short)0, acc[mt][nt], false, false);
  }
}

// C[M,N] = A[M,K] @ B[N,K]^T   (both operands K-contiguous = "NT" form)
// AF32: inputs are f32, converted to bf16 during a synchronous LDS fill.
// CF32: write f32 * scale; else write bf16.
template <bool AF32, bool CF32>
__global__ __launch_bounds__(256)
void gemm_nt(const void* __restrict__ Ap, const void* __restrict__ Bp,
             void* __restrict__ Cp, int M, int N, int Kdim,
             long long aBS, long long bBS, long long cBS, float scale) {
  __shared__ bf16_t sA[2][M_BLK * LDT];
  __shared__ bf16_t sB[2][N_BLK * LDT];
  const int bz  = blockIdx.z;
  const int m0  = blockIdx.y * M_BLK;
  const int n0  = blockIdx.x * N_BLK;
  const int tid = threadIdx.x;
  const int lane = tid & 31, wid = tid >> 5;
  const int wm = wid >> 1, wn = wid & 1;

  v8f acc[2][4] = {};

  if (AF32) {
    // --- synchronous fill, f32 -> bf16 fused, single buffer -----------------
    const float* A = (const float*)Ap + bz * aBS;
    const float* B = (const float*)Bp + bz * bBS;
    for (int k0 = 0; k0 < Kdim; k0 += K_BLK) {
      const int cg = tid & 15, row = tid >> 4;  // 16x4 f32 cols, 16 rows / pass
#pragma unroll
      for (int rr = 0; rr < 8; ++rr) {
        const int r = row + rr * 16;
        {
          v4f f = *(const v4f*)(A + (long long)(m0 + r) * Kdim + k0 + cg * 4);
          v4bf h; h[0] = (bf16_t)f[0]; h[1] = (bf16_t)f[1];
                  h[2] = (bf16_t)f[2]; h[3] = (bf16_t)f[3];
          *(v4bf*)&sA[0][r * LDT + cg * 4] = h;
        }
        {
          v4f f = *(const v4f*)(B + (long long)(n0 + r) * Kdim + k0 + cg * 4);
          v4bf h; h[0] = (bf16_t)f[0]; h[1] = (bf16_t)f[1];
                  h[2] = (bf16_t)f[2]; h[3] = (bf16_t)f[3];
          *(v4bf*)&sB[0][r * LDT + cg * 4] = h;
        }
      }
      if (k0 + K_BLK < Kdim) {
        __builtin_prefetch(A + (long long)(m0 + row) * Kdim + k0 + K_BLK, 0, 1);
        __builtin_prefetch(B + (long long)(n0 + row) * Kdim + k0 + K_BLK, 0, 1);
      }
      __syncthreads();
      mma_block(&sA[0][0], &sB[0][0], acc, wm, wn, lane);
      __syncthreads();
    }
  } else {
    const bf16_t* A = (const bf16_t*)Ap + bz * aBS;
    const bf16_t* B = (const bf16_t*)Bp + bz * bBS;
#if HAVE_ASYNC_LDS
    // --- async global->LDS copies (ASYNCcnt), double buffered ---------------
    // 128 rows x 8 16B-chunks per matrix = 1024 chunks -> 4 wave-instructions
    // per matrix per 256-thread block => 8 outstanding async ops per fill.
    auto fill_async = [&](int buf, int kk0) {
#pragma unroll
      for (int it = 0; it < 4; ++it) {
        const int c = tid + it * 256;
        const int row = c >> 3, colg = c & 7;
        __builtin_amdgcn_global_load_async_to_lds_b128(
            (g_v4i*)(A + (long long)(m0 + row) * Kdim + kk0 + colg * 8),
            (l_v4i*)&sA[buf][row * LDT + colg * 8],
            0, 0);
      }
#pragma unroll
      for (int it = 0; it < 4; ++it) {
        const int c = tid + it * 256;
        const int row = c >> 3, colg = c & 7;
        __builtin_amdgcn_global_load_async_to_lds_b128(
            (g_v4i*)(B + (long long)(n0 + row) * Kdim + kk0 + colg * 8),
            (l_v4i*)&sB[buf][row * LDT + colg * 8],
            0, 0);
      }
    };
    fill_async(0, 0);
    int p = 0;
    for (int k0 = 0; k0 < Kdim; k0 += K_BLK, p ^= 1) {
      const bool hasNext = (k0 + K_BLK) < Kdim;
      if (hasNext) fill_async(p ^ 1, k0 + K_BLK);  // prev compute on p^1 done
      if (hasNext) __builtin_amdgcn_s_wait_asynccnt(8);  // wait buffer p only
      else         __builtin_amdgcn_s_wait_asynccnt(0);
      __syncthreads();                              // all waves' fills landed
      mma_block(&sA[p][0], &sB[p][0], acc, wm, wn, lane);
      __syncthreads();                              // p reusable next iter
    }
#else
    // --- synchronous bf16 fill fallback -------------------------------------
    for (int k0 = 0; k0 < Kdim; k0 += K_BLK) {
      const int cg = tid & 7, row = tid >> 3;  // 8x8 bf16 cols, 32 rows / pass
#pragma unroll
      for (int rr = 0; rr < 4; ++rr) {
        const int r = row + rr * 32;
        *(v8bf*)&sA[0][r * LDT + cg * 8] =
            *(const v8bf*)(A + (long long)(m0 + r) * Kdim + k0 + cg * 8);
        *(v8bf*)&sB[0][r * LDT + cg * 8] =
            *(const v8bf*)(B + (long long)(n0 + r) * Kdim + k0 + cg * 8);
      }
      __syncthreads();
      mma_block(&sA[0][0], &sB[0][0], acc, wm, wn, lane);
      __syncthreads();
    }
#endif
  }

  // Epilogue: C/D layout (ISA 7.12.2): VGPR j, lane<16 -> (M=j, N=lane);
  // lane>=16 -> (M=8+j, N=lane-16).
  const int hi = lane >> 4, l = lane & 15;
#pragma unroll
  for (int mt = 0; mt < 2; ++mt)
#pragma unroll
    for (int nt = 0; nt < 4; ++nt)
#pragma unroll
      for (int j = 0; j < 8; ++j) {
        const int m = m0 + wm * 32 + mt * 16 + hi * 8 + j;
        const int n = n0 + wn * 64 + nt * 16 + l;
        if (CF32) {
          float* C = (float*)Cp + bz * cBS;
          C[(long long)m * N + n] = acc[mt][nt][j] * scale;
        } else {
          bf16_t* C = (bf16_t*)Cp + bz * cBS;
          C[(long long)m * N + n] = (bf16_t)acc[mt][nt][j];
        }
      }
}

// N[b][m,n] = sum_s V[b][s,m] * K[b][s,n]   (i.e. N = V^T K = (K^T V)^T)
// Both operands enter LDS through a transpose-on-store fill.
__global__ __launch_bounds__(256)
void gemm_tn(const bf16_t* __restrict__ Vp, const bf16_t* __restrict__ Kp,
             bf16_t* __restrict__ Np, int S, int D) {
  __shared__ bf16_t sA[M_BLK * LDT];  // sA[m][s] = V[s][m]
  __shared__ bf16_t sB[N_BLK * LDT];  // sB[n][s] = K[s][n]
  const int bz = blockIdx.z;
  const bf16_t* V  = Vp + (long long)bz * S * D;
  const bf16_t* Kb = Kp + (long long)bz * S * D;
  bf16_t*       Nb = Np + (long long)bz * D * D;
  const int m0 = blockIdx.y * M_BLK;
  const int n0 = blockIdx.x * N_BLK;
  const int tid = threadIdx.x, lane = tid & 31, wid = tid >> 5;
  const int wm = wid >> 1, wn = wid & 1;

  v8f acc[2][4] = {};

  for (int s0 = 0; s0 < S; s0 += K_BLK) {
    const int cg = tid & 15, srow = tid >> 4;  // 16 col-groups of 8, 16 s-rows
#pragma unroll
    for (int rr = 0; rr < 4; ++rr) {
      const int s = srow + rr * 16;
      v8bf hv = *(const v8bf*)(V  + (long long)(s0 + s) * D + m0 + cg * 8);
      v8bf hk = *(const v8bf*)(Kb + (long long)(s0 + s) * D + n0 + cg * 8);
#pragma unroll
      for (int i = 0; i < 8; ++i) {
        sA[(cg * 8 + i) * LDT + s] = hv[i];
        sB[(cg * 8 + i) * LDT + s] = hk[i];
      }
    }
    if (s0 + K_BLK < S) {
      __builtin_prefetch(V  + (long long)(s0 + K_BLK + srow) * D + m0, 0, 1);
      __builtin_prefetch(Kb + (long long)(s0 + K_BLK + srow) * D + n0, 0, 1);
    }
    __syncthreads();
    mma_block(sA, sB, acc, wm, wn, lane);
    __syncthreads();
  }

  const int hi = lane >> 4, l = lane & 15;
#pragma unroll
  for (int mt = 0; mt < 2; ++mt)
#pragma unroll
    for (int nt = 0; nt < 4; ++nt)
#pragma unroll
      for (int j = 0; j < 8; ++j) {
        const int m = m0 + wm * 32 + mt * 16 + hi * 8 + j;
        const int n = n0 + wn * 64 + nt * 16 + l;
        Nb[(long long)m * D + n] = (bf16_t)acc[mt][nt][j];
      }
}

// Memory-bound one-pass f32 -> bf16 conversion (~2 us at 23.3 TB/s for 50 MB).
__global__ __launch_bounds__(256)
void cvt_f32_bf16(const float* __restrict__ src, bf16_t* __restrict__ dst,
                  long long n) {
  const long long i = ((long long)blockIdx.x * 256 + threadIdx.x) * 4;
  if (i + 3 < n) {
    v4f f = *(const v4f*)(src + i);
    v4bf h; h[0] = (bf16_t)f[0]; h[1] = (bf16_t)f[1];
            h[2] = (bf16_t)f[2]; h[3] = (bf16_t)f[3];
    *(v4bf*)(dst + i) = h;
  }
}

extern "C" void kernel_launch(void* const* d_in, const int* in_sizes, int n_in,
                              void* d_out, int out_size, void* d_ws, size_t ws_size,
                              hipStream_t stream) {
  const float* x  = (const float*)d_in[0];  // [4,2048,1024]
  const float* Wq = (const float*)d_in[1];  // [1024,1024]
  const float* Wk = (const float*)d_in[2];
  const float* Wv = (const float*)d_in[3];
  float* out = (float*)d_out;               // [4,2048,1024] f32

  const int B = 4, S = 2048, D = 1024;
  const long long MS = (long long)B * S;    // 8192 fused rows
  const long long XE = MS * D;              // x / Q / K / V element count
  const long long WE = (long long)D * D;    // one weight / one N[b]

  dim3 blk(256, 1, 1);
  dim3 g1(D / N_BLK, (unsigned)(MS / M_BLK), 1);
  dim3 g2(D / N_BLK, D / M_BLK, B);
  dim3 g3(D / N_BLK, S / M_BLK, B);

  // Preferred layout (all-bf16 GEMMs, async LDS fills):
  //   Xbf | Wq | Wk | Wv | Q | K | V | N   = ~82 MB
  const size_t needA = (size_t)(XE + 3 * WE + 3 * XE + B * WE) * sizeof(bf16_t);

  if (ws_size >= needA) {
    bf16_t* Xb  = (bf16_t*)d_ws;
    bf16_t* Wqb = Xb  + XE;
    bf16_t* Wkb = Wqb + WE;
    bf16_t* Wvb = Wkb + WE;
    bf16_t* Q   = Wvb + WE;
    bf16_t* Kq  = Q   + XE;
    bf16_t* Vq  = Kq  + XE;
    bf16_t* Nq  = Vq  + XE;

    hipLaunchKernelGGL(cvt_f32_bf16, dim3((unsigned)(XE / 1024)), blk, 0, stream, x,  Xb,  XE);
    hipLaunchKernelGGL(cvt_f32_bf16, dim3((unsigned)(WE / 1024)), blk, 0, stream, Wq, Wqb, WE);
    hipLaunchKernelGGL(cvt_f32_bf16, dim3((unsigned)(WE / 1024)), blk, 0, stream, Wk, Wkb, WE);
    hipLaunchKernelGGL(cvt_f32_bf16, dim3((unsigned)(WE / 1024)), blk, 0, stream, Wv, Wvb, WE);

    hipLaunchKernelGGL((gemm_nt<false, false>), g1, blk, 0, stream,
                       (const void*)Xb, (const void*)Wqb, (void*)Q,
                       (int)MS, D, D, 0LL, 0LL, 0LL, 1.0f);
    hipLaunchKernelGGL((gemm_nt<false, false>), g1, blk, 0, stream,
                       (const void*)Xb, (const void*)Wkb, (void*)Kq,
                       (int)MS, D, D, 0LL, 0LL, 0LL, 1.0f);
    hipLaunchKernelGGL((gemm_nt<false, false>), g1, blk, 0, stream,
                       (const void*)Xb, (const void*)Wvb, (void*)Vq,
                       (int)MS, D, D, 0LL, 0LL, 0LL, 1.0f);

    hipLaunchKernelGGL(gemm_tn, g2, blk, 0, stream, Vq, Kq, Nq, S, D);

    hipLaunchKernelGGL((gemm_nt<false, true>), g3, blk, 0, stream,
                       (const void*)Q, (const void*)Nq, (void*)out,
                       S, D, D, (long long)S * D, (long long)D * D,
                       (long long)S * D, 0.03125f /* 1/sqrt(1024) */);
  } else {
    // Fallback (~59 MB ws): fuse f32->bf16 into the stage-1 fills.
    bf16_t* Q  = (bf16_t*)d_ws;
    bf16_t* Kq = Q  + XE;
    bf16_t* Vq = Kq + XE;
    bf16_t* Nq = Vq + XE;

    hipLaunchKernelGGL((gemm_nt<true, false>), g1, blk, 0, stream,
                       (const void*)x, (const void*)Wq, (void*)Q,
                       (int)MS, D, D, 0LL, 0LL, 0LL, 1.0f);
    hipLaunchKernelGGL((gemm_nt<true, false>), g1, blk, 0, stream,
                       (const void*)x, (const void*)Wk, (void*)Kq,
                       (int)MS, D, D, 0LL, 0LL, 0LL, 1.0f);
    hipLaunchKernelGGL((gemm_nt<true, false>), g1, blk, 0, stream,
                       (const void*)x, (const void*)Wv, (void*)Vq,
                       (int)MS, D, D, 0LL, 0LL, 0LL, 1.0f);

    hipLaunchKernelGGL(gemm_tn, g2, blk, 0, stream, Vq, Kq, Nq, S, D);

    hipLaunchKernelGGL((gemm_nt<false, true>), g3, blk, 0, stream,
                       (const void*)Q, (const void*)Nq, (void*)out,
                       S, D, D, (long long)S * D, (long long)D * D,
                       (long long)S * D, 0.03125f);
  }
}
